// IMECClassifier_56427280335066
// MI455X (gfx1250) — compile-verified, matrix operand order checked by reading, not run on hardware
//
#include <hip/hip_runtime.h>

// Problem constants (from the reference)
#define DB 256      // batch
#define DH 1024     // hidden
#define DL 10000    // labels
#define DD 8        // expert hidden
#define KCHUNK 256  // K staged per LDS round
#define LDSTRIDE (KCHUNK + 8)  // pad: col stride 528B -> bank stride 4, conflict-free b128 reads

typedef __attribute__((ext_vector_type(16))) __bf16 v16bf;
typedef __attribute__((ext_vector_type(8)))  __bf16 v8bf;
typedef __attribute__((ext_vector_type(4)))  __bf16 v4bf;
typedef __attribute__((ext_vector_type(8)))  float  v8f;
typedef __attribute__((ext_vector_type(4)))  int    v4i;

#if defined(__AMDGCN__) && __has_builtin(__builtin_amdgcn_global_load_async_to_lds_b128)
#define HAVE_ASYNC 1
typedef __attribute__((address_space(1))) v4i as1_v4i;   // global int4
typedef __attribute__((address_space(3))) v4i as3_v4i;   // LDS int4
#if __has_builtin(__builtin_amdgcn_s_wait_asynccnt)
#define ASYNC_WAIT() __builtin_amdgcn_s_wait_asynccnt(0)
#else
#define ASYNC_WAIT() asm volatile("s_wait_asynccnt 0" ::: "memory")
#endif
#else
#define HAVE_ASYNC 0
#endif

// Native fp32->bf16 (lets clang select v_cvt_pk_bf16_f32)
__device__ __forceinline__ __bf16 f2bf(float f) { return (__bf16)f; }

// XOR lane swap within wave32 via ds_swizzle (group-of-32: and=0x1F, or=0, xor=M)
template <int M>
__device__ __forceinline__ float xor_swz(float x) {
  int xi = __builtin_bit_cast(int, x);
  xi = __builtin_amdgcn_ds_swizzle(xi, 0x1F | (M << 10));
  return __builtin_bit_cast(float, xi);
}

// ---------------------------------------------------------------------------
// Pass 1: one-shot fp32 -> bf16 conversion of embeddings into d_ws (512 KB).
// 256 blocks x 256 threads x 4 elements = 256*1024 elements.
// ---------------------------------------------------------------------------
__global__ __launch_bounds__(256)
void emb_to_bf16_kernel(const float* __restrict__ in, __bf16* __restrict__ outb) {
  const int i = (blockIdx.x * 256 + threadIdx.x) * 4;
  const float4 v = *(const float4*)(in + i);
  v4bf o;
  o[0] = f2bf(v.x); o[1] = f2bf(v.y); o[2] = f2bf(v.z); o[3] = f2bf(v.w);
  *(v4bf*)(outb + i) = o;
}

// ---------------------------------------------------------------------------
// Pass 2: fused per-label MLP. One workgroup (8 wave32) per label pair.
// ---------------------------------------------------------------------------
__global__ __launch_bounds__(256)
void imec_fused_mlp_kernel(const __bf16* __restrict__ embB,   // [256][1024] bf16
                           const float* __restrict__ W1,
                           const float* __restrict__ b1,
                           const float* __restrict__ W2,
                           const float* __restrict__ b2,
                           float* __restrict__ out) {
#if HAVE_ASYNC
  __shared__ __align__(16) float lds_raw[2 * KCHUNK * DD];       // 16 KB staged fp32 W1 chunk
#endif
  __shared__ __align__(16) __bf16 lds_w[16 * LDSTRIDE];          // transposed bf16 B-panel

  const int tid  = threadIdx.x;
  const int lane = tid & 31;
  const int wave = tid >> 5;

  const int l0 = blockIdx.x * 2;                    // label pair
  const float* gW1 = W1 + (size_t)l0 * DH * DD;     // 2 labels x [1024,8] row-major

  // Lane geometry shared by B-fragment and C/D epilogue
  const int ncol     = lane & 15;    // B/C column: 0..7 -> label l0, 8..15 -> l0+1
  const int laneHalf = lane >> 4;
  const int dsel     = ncol & 7;
  const int label    = l0 + (ncol >> 3);
  const int klane    = laneHalf * 8; // per-lane K sub-offset

  const float b1v = b1[l0 * DD + ncol];
  const float w2v = W2[l0 * DD + ncol];
  const float b2v = b2[label];

  // Each wave owns two 16-row M-tiles: rows [wave*16) and [(wave+8)*16)
  const int m0 = wave * 16;
  const int m1 = (wave + 8) * 16;
  const int arow = lane & 15;
  const __bf16* embA0 = embB + (size_t)(m0 + arow) * DH + klane;
  const __bf16* embA1 = embB + (size_t)(m1 + arow) * DH + klane;

  v8f acc0 = {};
  v8f acc1 = {};

  for (int h0 = 0; h0 < DH; h0 += KCHUNK) {
    __syncthreads();  // previous chunk fully consumed before restaging

#if HAVE_ASYNC
    // Async DMA: 2*256*8 fp32 (16 KB) -> LDS, 4 b128 per thread (ASYNCcnt path)
#pragma unroll
    for (int i = 0; i < 4; ++i) {
      const int f4   = tid + i * 256;       // float4 id within chunk
      const int e    = f4 * 4;              // float id within chunk
      const int loff = e >> 11;             // which label of the pair
      const int gidx = e + loff * 6144 + h0 * DD;
      __builtin_amdgcn_global_load_async_to_lds_b128(
          (as1_v4i*)(gW1 + gidx), (as3_v4i*)(&lds_raw[e]), 0, 0);
    }
    ASYNC_WAIT();
#endif
    __syncthreads();

    // Transpose + fp32->bf16: lds_w[col][k] with col = (label-in-pair)*8 + d
#pragma unroll
    for (int i = 0; i < 4; ++i) {
      const int f4   = tid + i * 256;
      const int e    = f4 * 4;
      const int loff = e >> 11;
      const int r    = e & 2047;
      const int hh   = r >> 3;              // chunk-local k
      const int colb = (loff << 3) + (r & 7);
#if HAVE_ASYNC
      const float4 v = *(const float4*)(&lds_raw[e]);
#else
      const float4 v = *(const float4*)(gW1 + e + loff * 6144 + h0 * DD);
#endif
      lds_w[(colb + 0) * LDSTRIDE + hh] = f2bf(v.x);
      lds_w[(colb + 1) * LDSTRIDE + hh] = f2bf(v.y);
      lds_w[(colb + 2) * LDSTRIDE + hh] = f2bf(v.z);
      lds_w[(colb + 3) * LDSTRIDE + hh] = f2bf(v.w);
    }
    __syncthreads();

    // 8 WMMA K-steps per chunk, 2 M-tiles per wave sharing one B-fragment.
    // All operands are bf16 already: pure b128 loads + WMMA, no conversion VALU.
#pragma unroll 2
    for (int kc = 0; kc < KCHUNK; kc += 32) {
      const __bf16* bp = &lds_w[ncol * LDSTRIDE + kc + klane];
      const v8bf bl = *(const v8bf*)(bp);        // K kb..kb+7   (ds_load_b128)
      const v8bf bh = *(const v8bf*)(bp + 16);   // K kb+16..+23 (ds_load_b128)
      const v16bf bfrag = __builtin_shufflevector(
          bl, bh, 0, 1, 2, 3, 4, 5, 6, 7, 8, 9, 10, 11, 12, 13, 14, 15);

      const __bf16* a0p = embA0 + h0 + kc;
      const v8bf a0l = *(const v8bf*)(a0p);      // global_load_b128
      const v8bf a0h = *(const v8bf*)(a0p + 16);
      const v16bf a0 = __builtin_shufflevector(
          a0l, a0h, 0, 1, 2, 3, 4, 5, 6, 7, 8, 9, 10, 11, 12, 13, 14, 15);

      const __bf16* a1p = embA1 + h0 + kc;
      const v8bf a1l = *(const v8bf*)(a1p);
      const v8bf a1h = *(const v8bf*)(a1p + 16);
      const v16bf a1 = __builtin_shufflevector(
          a1l, a1h, 0, 1, 2, 3, 4, 5, 6, 7, 8, 9, 10, 11, 12, 13, 14, 15);

      acc0 = __builtin_amdgcn_wmma_f32_16x16x32_bf16(
          false, a0, false, bfrag, (short)0, acc0, false, false);
      acc1 = __builtin_amdgcn_wmma_f32_16x16x32_bf16(
          false, a1, false, bfrag, (short)0, acc1, false, false);
    }
  }

  // Epilogue: ReLU, scale by W2, reduce over d (8 lanes) via ds_swizzle XOR, +b2.
  // C layout: VGPR i, lanes 0-15 -> M=i, lanes 16-31 -> M=i+8, N=ncol.
  auto epilogue = [&](const v8f& acc, int mbase) {
#pragma unroll
    for (int i = 0; i < 8; ++i) {
      float h = acc[i] + b1v;
      h = fmaxf(h, 0.0f);
      float t = h * w2v;
      t += xor_swz<1>(t);
      t += xor_swz<2>(t);
      t += xor_swz<4>(t);
      if (dsel == 0) {
        const int m = mbase + i + laneHalf * 8;
        out[(size_t)m * DL + label] = t + b2v;
      }
    }
  };
  epilogue(acc0, m0);
  epilogue(acc1, m1);
}

extern "C" void kernel_launch(void* const* d_in, const int* in_sizes, int n_in,
                              void* d_out, int out_size, void* d_ws, size_t ws_size,
                              hipStream_t stream) {
  (void)in_sizes; (void)n_in; (void)out_size; (void)ws_size;
  const float* emb = (const float*)d_in[0];
  const float* W1  = (const float*)d_in[1];
  const float* b1  = (const float*)d_in[2];
  const float* W2  = (const float*)d_in[3];
  const float* b2  = (const float*)d_in[4];
  float* out = (float*)d_out;
  __bf16* embB = (__bf16*)d_ws;   // 256*1024*2 = 512 KB scratch

  // Pass 1: embeddings fp32 -> bf16 (runs every call; stream-ordered)
  emb_to_bf16_kernel<<<dim3(DB * DH / (256 * 4)), dim3(256), 0, stream>>>(emb, embB);

  // Pass 2: fused MLP, one workgroup per label pair
  imec_fused_mlp_kernel<<<dim3(DL / 2), dim3(256), 0, stream>>>(embB, W1, b1, W2, b2, out);
}
